// GSNN_84670985273779
// MI455X (gfx1250) — matrix operand order repaired and checked by reading, not compiled, not run on hardware
//
#include <hip/hip_runtime.h>
#include <hip/hip_bf16.h>

typedef __attribute__((ext_vector_type(16))) _Float16 v16h;
typedef __attribute__((ext_vector_type(8)))  _Float16 v8h;
typedef __attribute__((ext_vector_type(8)))  float    v8f;

// Problem constants (from reference)
static constexpr int NN     = 16384;
static constexpr int NFEAT  = 300;
static constexpr int NFEATP = 320;   // padded to multiple of 32
static constexpr int NHID   = 512;
static constexpr int NOUT   = 256;
static constexpr int NACT   = 64;
static constexpr int NEIGH  = 2048;
static constexpr int S      = NACT + NEIGH;   // 2112 = 132*16 = 66*32

// ---------------- prep kernels ----------------
__global__ void k_cur(const int* __restrict__ act, const int* __restrict__ nb,
                      int* __restrict__ cur) {
    int i = blockIdx.x * 256 + threadIdx.x;
    if (i < S) cur[i] = (i < NACT) ? act[i] : nb[i - NACT];
}

__global__ void k_buildA(const float* __restrict__ adj, const int* __restrict__ cur,
                         _Float16* __restrict__ Ah) {
    long long idx = (long long)blockIdx.x * 256 + threadIdx.x;
    if (idx >= (long long)S * S) return;
    int i = (int)(idx / S);
    int j = (int)(idx % S);
    Ah[idx] = (_Float16)adj[(size_t)cur[i] * NN + cur[j]];
}

__global__ void k_buildX(const float* __restrict__ x, const int* __restrict__ cur,
                         _Float16* __restrict__ Xh) {
    int idx = blockIdx.x * 256 + threadIdx.x;
    if (idx >= S * NFEATP) return;
    int i = idx / NFEATP, k = idx % NFEATP;
    Xh[idx] = (k < NFEAT) ? (_Float16)x[(size_t)cur[i] * NFEAT + k] : (_Float16)0.f;
}

__global__ void k_buildW1T(const float* __restrict__ W1, _Float16* __restrict__ W1T) {
    int idx = blockIdx.x * 256 + threadIdx.x;           // [NHID][NFEATP]
    if (idx >= NHID * NFEATP) return;
    int n = idx / NFEATP, k = idx % NFEATP;
    W1T[idx] = (k < NFEAT) ? (_Float16)W1[(size_t)k * NHID + n] : (_Float16)0.f;
}

__global__ void k_buildW2T(const float* __restrict__ W2, _Float16* __restrict__ W2T) {
    int idx = blockIdx.x * 256 + threadIdx.x;           // [NOUT][NHID]
    if (idx >= NOUT * NHID) return;
    int n = idx / NHID, k = idx % NHID;
    W2T[idx] = (_Float16)W2[(size_t)k * NOUT + n];
}

// ---------------- WMMA GEMM ----------------
// C[MM,NC] = A[MM,KK](f16 row-major) * B, with B given TRANSPOSED: BT[NC,KK] f16
// row-major. One wave computes a 16x64 tile (4 wmma n-tiles), K-loop is register
// double-buffered so wmmas of slab t overlap the loads of slab t+1.
// STORE_MODE: 0 = transposed f16 (DT[NC][MM]); 1 = row-major f16; 2 = row-major f32.
template <int STORE_MODE, bool RELU, int MM, int NC, int KK>
__global__ void wmma_gemm(const _Float16* __restrict__ A, const _Float16* __restrict__ BT,
                          void* __restrict__ D) {
    constexpr int nGroups = NC >> 6;             // 64 columns per wave
    constexpr int mTiles  = MM >> 4;
    constexpr int NK      = KK >> 5;             // K-slabs of 32
    const int lane = threadIdx.x & 31;
    const int wave = blockIdx.x * (blockDim.x >> 5) + (threadIdx.x >> 5);
    if (wave >= mTiles * nGroups) return;
    const int mTile  = wave / nGroups;
    const int nGroup = wave % nGroups;
    const int m0 = mTile << 4;
    const int n0 = nGroup << 6;
    const int mr = lane & 15;                    // row within A tile / col within B tile
    const int hh = lane >> 4;                    // half-wave select

    v8f acc[4] = {};

    // A fragment: lane (mr,hh) holds K = hh*8 + e (e<8) and 16 + hh*8 + (e-8).
    const _Float16* aBase = A + (size_t)(m0 + mr) * KK + hh * 8;
    // B fragment: lane (n=mr, hh) holds K = hh*8 + e (e<8) and 16 + hh*8 + (e-8).
    const _Float16* bBase[4];
#pragma unroll
    for (int j = 0; j < 4; ++j)
        bBase[j] = BT + (size_t)(n0 + j * 16 + mr) * KK + hh * 8;

    v8h Alo[2], Ahi[2], Bl[2][4], Bh[2][4];

    auto issue = [&](int buf, int kb) {
        Alo[buf] = *(const v8h*)(aBase + kb);
        Ahi[buf] = *(const v8h*)(aBase + kb + 16);
#pragma unroll
        for (int j = 0; j < 4; ++j) {
            Bl[buf][j] = *(const v8h*)(bBase[j] + kb);
            Bh[buf][j] = *(const v8h*)(bBase[j] + kb + 16);
        }
    };
    auto compute = [&](int buf) {
        v16h af;
#pragma unroll
        for (int e = 0; e < 8; ++e) { af[e] = Alo[buf][e]; af[e + 8] = Ahi[buf][e]; }
#pragma unroll
        for (int j = 0; j < 4; ++j) {
            v16h bf;
#pragma unroll
            for (int e = 0; e < 8; ++e) { bf[e] = Bl[buf][j][e]; bf[e + 8] = Bh[buf][j][e]; }
            acc[j] = __builtin_amdgcn_wmma_f32_16x16x32_f16(
                false, af, false, bf, (short)0, acc[j], false, false);
        }
    };

    issue(0, 0);
#pragma unroll 2
    for (int t = 0; t < NK; ++t) {
        if (t + 1 < NK) issue((t + 1) & 1, (t + 1) * 32);
        compute(t & 1);
    }

    // C/D layout: element r of acc -> row m0 + r + 8*hh, col n0 + j*16 + mr.
    if (STORE_MODE == 0) {
        _Float16* DT = (_Float16*)D;
#pragma unroll
        for (int j = 0; j < 4; ++j) {
            v8h o;
#pragma unroll
            for (int r = 0; r < 8; ++r) {
                float t = acc[j][r];
                if (RELU) t = t > 0.f ? t : 0.f;
                o[r] = (_Float16)t;
            }
            // rows m0+8*hh .. m0+8*hh+7 are contiguous in DT's column direction
            *(v8h*)(DT + (size_t)(n0 + j * 16 + mr) * MM + m0 + hh * 8) = o;
        }
    } else if (STORE_MODE == 1) {
        _Float16* Dh = (_Float16*)D;
#pragma unroll
        for (int j = 0; j < 4; ++j)
#pragma unroll
            for (int r = 0; r < 8; ++r) {
                float t = acc[j][r];
                if (RELU) t = t > 0.f ? t : 0.f;
                Dh[(size_t)(m0 + r + 8 * hh) * NC + n0 + j * 16 + mr] = (_Float16)t;
            }
    } else {
        float* Df = (float*)D;
#pragma unroll
        for (int j = 0; j < 4; ++j)
#pragma unroll
            for (int r = 0; r < 8; ++r) {
                float t = acc[j][r];
                if (RELU) t = t > 0.f ? t : 0.f;
                Df[(size_t)(m0 + r + 8 * hh) * NC + n0 + j * 16 + mr] = t;
            }
    }
}

// ---------------- tail kernels ----------------
__global__ void k_dot_v(const float* __restrict__ H2, const float* __restrict__ imp_w,
                        float* __restrict__ v) {
    __shared__ float red[256];
    int i = blockIdx.x, t = threadIdx.x;
    red[t] = H2[(size_t)i * NOUT + t] * imp_w[t];
    __syncthreads();
    for (int o = 128; o > 0; o >>= 1) { if (t < o) red[t] += red[t + o]; __syncthreads(); }
    if (t == 0) v[i] = red[0];
}

__global__ void k_imp(const _Float16* __restrict__ Ah, const float* __restrict__ v,
                      const float* __restrict__ imp_b, float* __restrict__ imp) {
    __shared__ float red[256];
    int i = blockIdx.x, t = threadIdx.x;
    float s = 0.f;
    for (int j = t; j < S; j += 256) s += (float)Ah[(size_t)i * S + j] * v[j];
    red[t] = s;
    __syncthreads();
    for (int o = 128; o > 0; o >>= 1) { if (t < o) red[t] += red[t + o]; __syncthreads(); }
    if (t == 0) imp[i] = red[0] + imp_b[0];
}

__global__ void k_final(const float* __restrict__ imp, const int* __restrict__ nb,
                        float* __restrict__ out) {
    __shared__ float red[256];
    __shared__ float denom;
    int t = threadIdx.x;
    float s = 0.f;
    for (int j = t; j < S; j += 256) s += fabsf(imp[j]);
    red[t] = s;
    __syncthreads();
    for (int o = 128; o > 0; o >>= 1) { if (t < o) red[t] += red[t + o]; __syncthreads(); }
    if (t == 0) denom = fmaxf(red[0], 1e-12f);
    __syncthreads();
    for (int j = t; j < NEIGH; j += 256) {
        out[j]         = imp[NACT + j] / denom;
        out[NEIGH + j] = (float)nb[j];
    }
}

// ---------------- launcher ----------------
extern "C" void kernel_launch(void* const* d_in, const int* in_sizes, int n_in,
                              void* d_out, int out_size, void* d_ws, size_t ws_size,
                              hipStream_t stream) {
    const float* x    = (const float*)d_in[0];
    const float* adj  = (const float*)d_in[1];
    const int*   act  = (const int*)d_in[2];
    const int*   nb   = (const int*)d_in[3];
    const float* W1   = (const float*)d_in[4];
    const float* W2   = (const float*)d_in[5];
    const float* impw = (const float*)d_in[6];
    const float* impb = (const float*)d_in[7];
    float*       out  = (float*)d_out;

    char* ws = (char*)d_ws;
    size_t off = 0;
    auto alloc = [&](size_t bytes) -> char* {
        off = (off + 255) & ~(size_t)255;
        char* p = ws + off;
        off += bytes;
        return p;
    };
    int*       cur  = (int*)alloc((size_t)S * 4);
    _Float16*  Ah   = (_Float16*)alloc((size_t)S * S * 2);
    _Float16*  Xh   = (_Float16*)alloc((size_t)S * NFEATP * 2);
    _Float16*  W1T  = (_Float16*)alloc((size_t)NHID * NFEATP * 2);
    _Float16*  W2T  = (_Float16*)alloc((size_t)NOUT * NHID * 2);
    _Float16*  T1T  = (_Float16*)alloc((size_t)NHID * S * 2);
    _Float16*  H1   = (_Float16*)alloc((size_t)S * NHID * 2);
    _Float16*  T2T  = (_Float16*)alloc((size_t)NOUT * S * 2);
    float*     H2   = (float*)alloc((size_t)S * NOUT * 4);
    float*     v    = (float*)alloc((size_t)S * 4);
    float*     imp  = (float*)alloc((size_t)S * 4);
    (void)in_sizes; (void)n_in; (void)out_size; (void)ws_size;

    // gather + convert
    k_cur<<<(S + 255) / 256, 256, 0, stream>>>(act, nb, cur);
    k_buildA<<<(int)(((long long)S * S + 255) / 256), 256, 0, stream>>>(adj, cur, Ah);
    k_buildX<<<(S * NFEATP + 255) / 256, 256, 0, stream>>>(x, cur, Xh);
    k_buildW1T<<<(NHID * NFEATP + 255) / 256, 256, 0, stream>>>(W1, W1T);
    k_buildW2T<<<(NOUT * NHID + 255) / 256, 256, 0, stream>>>(W2, W2T);

    auto gblocks = [](int M, int Nc) { return ((M >> 4) * (Nc >> 6) + 7) / 8; };

    // T1^T = (X @ W1)^T                      [NHID][S]
    wmma_gemm<0, false, S, NHID, NFEATP>
        <<<gblocks(S, NHID), 256, 0, stream>>>(Xh, W1T, T1T);
    // H1 = relu(A @ T1)  row-major f16       [S][NHID]
    wmma_gemm<1, true, S, NHID, S>
        <<<gblocks(S, NHID), 256, 0, stream>>>(Ah, T1T, H1);
    // T2^T = (H1 @ W2)^T                     [NOUT][S]
    wmma_gemm<0, false, S, NOUT, NHID>
        <<<gblocks(S, NOUT), 256, 0, stream>>>(H1, W2T, T2T);
    // H2 = A @ T2  row-major f32             [S][NOUT]
    wmma_gemm<2, false, S, NOUT, S>
        <<<gblocks(S, NOUT), 256, 0, stream>>>(Ah, T2T, H2);

    // imp = A @ (H2 @ imp_w) + b ; L1-normalize ; emit outputs
    k_dot_v<<<S, 256, 0, stream>>>(H2, impw, v);
    k_imp<<<S, 256, 0, stream>>>(Ah, v, impb, imp);
    k_final<<<1, 256, 0, stream>>>(imp, nb, out);
}
